// _AttentionLayer_15874199126656
// MI455X (gfx1250) — compile-verified, matrix operand order checked by reading, not run on hardware
//
#include <hip/hip_runtime.h>

typedef unsigned short ushort_t;

typedef __attribute__((ext_vector_type(16))) __bf16        bf16x16;
typedef __attribute__((ext_vector_type(8)))  float         f32x8;
typedef __attribute__((ext_vector_type(4)))  float         f32x4;
typedef __attribute__((ext_vector_type(4)))  unsigned int  u32x4;
typedef __attribute__((ext_vector_type(2)))  unsigned int  u32x2;

union Frag {
    bf16x16 f;
    u32x4   q[2];
};

__device__ __forceinline__ ushort_t f2bf(float x) {
    unsigned int u = __float_as_uint(x);
    u += 0x7FFFu + ((u >> 16) & 1u);          // round-to-nearest-even
    return (ushort_t)(u >> 16);
}

__device__ __forceinline__ f32x8 wmma_bf16(Frag a, Frag b, f32x8 c) {
    // v_wmma_f32_16x16x32_bf16: D = A(16x32) x B(32x16) + C(16x16 f32)
    return __builtin_amdgcn_wmma_f32_16x16x32_bf16(
        /*neg_a=*/false, a.f, /*neg_b=*/false, b.f,
        /*c_mod=*/(short)0, c, /*reuse_a=*/false, /*reuse_b=*/false);
}

// ---------------------------------------------------------------------------
// Tiled WMMA GEMM: C[M,N] = A[M,K] * W[K,N] + bias
// Block tile 128x128, BK=32, 256 threads = 8 wave32 waves (4x2 wave grid),
// each wave computes a 32x64 sub-tile (2x4 WMMA tiles, f32 accum).
// Register-staged double-buffered LDS: next tile's global loads issue before
// the WMMA block, LDS stores land in the alternate buffer, ONE barrier/step.
// ---------------------------------------------------------------------------
template <bool A_BF16, bool OUT_F32>
__global__ __launch_bounds__(256)
void gemm_wmma(const void* __restrict__ Aptr, const float* __restrict__ W,
               const float* __restrict__ bias, void* __restrict__ Cptr,
               int M, int N, int K)
{
    // +8 half padding: row stride 80B (5x16B) -> conflict-free b128 reads
    __shared__ ushort_t As [2][128 * 40];
    __shared__ ushort_t BsT[2][128 * 40];

    const int tid  = threadIdx.x;
    const int wave = tid >> 5, lane = tid & 31;
    const int ln = lane & 15, hi = lane >> 4;
    const int k0l = hi * 8;                    // ISA 16-bit A/B fragment K-offset
    const int wm0 = (wave >> 1) * 32;
    const int wn0 = (wave & 1) * 64;
    const int bm  = blockIdx.y * 128;
    const int bn  = blockIdx.x * 128;

    f32x4 aReg[4];        // f32 A staging
    u32x4 abReg[2];       // bf16 A staging
    f32x4 wReg[4];        // W staging (always f32 source)

    auto globalLoad = [&](int kt) {
        if constexpr (A_BF16) {
            const ushort_t* Ab = (const ushort_t*)Aptr;
#pragma unroll
            for (int t = 0; t < 2; t++) {
                int cid = tid + t * 256;            // 512 x 16B chunks
                int r = cid >> 2, c8 = (cid & 3) * 8;
                abReg[t] = *(const u32x4*)&Ab[(size_t)(bm + r) * K + kt + c8];
            }
        } else {
            const float* Af = (const float*)Aptr;
#pragma unroll
            for (int t = 0; t < 4; t++) {
                int cid = tid + t * 256;            // 1024 x float4
                int r = cid >> 3, c4 = (cid & 7) * 4;
                aReg[t] = *(const f32x4*)&Af[(size_t)(bm + r) * K + kt + c4];
            }
        }
#pragma unroll
        for (int t = 0; t < 4; t++) {
            int cid = tid + t * 256;
            int r  = cid >> 5;                     // k row 0..31
            int n4 = (cid & 31) * 4;               // n col 0..124
            wReg[t] = *(const f32x4*)&W[(size_t)(kt + r) * N + bn + n4];
        }
    };

    auto ldsStore = [&](int buf) {
        ushort_t* as = As[buf];
        ushort_t* bs = BsT[buf];
        if constexpr (A_BF16) {
#pragma unroll
            for (int t = 0; t < 2; t++) {
                int cid = tid + t * 256;
                int r = cid >> 2, c8 = (cid & 3) * 8;
                *(u32x4*)&as[r * 40 + c8] = abReg[t];
            }
        } else {
#pragma unroll
            for (int t = 0; t < 4; t++) {
                int cid = tid + t * 256;
                int r = cid >> 3, c4 = (cid & 7) * 4;
                u32x2 p;
                p.x = (unsigned)f2bf(aReg[t].x) | ((unsigned)f2bf(aReg[t].y) << 16);
                p.y = (unsigned)f2bf(aReg[t].z) | ((unsigned)f2bf(aReg[t].w) << 16);
                *(u32x2*)&as[r * 40 + c4] = p;
            }
        }
#pragma unroll
        for (int t = 0; t < 4; t++) {
            int cid = tid + t * 256;
            int r  = cid >> 5;
            int n4 = (cid & 31) * 4;
            bs[(n4 + 0) * 40 + r] = f2bf(wReg[t].x);
            bs[(n4 + 1) * 40 + r] = f2bf(wReg[t].y);
            bs[(n4 + 2) * 40 + r] = f2bf(wReg[t].z);
            bs[(n4 + 3) * 40 + r] = f2bf(wReg[t].w);
        }
    };

    f32x8 acc[2][4];
#pragma unroll
    for (int i = 0; i < 2; i++)
#pragma unroll
        for (int j = 0; j < 4; j++) acc[i][j] = (f32x8)0.0f;

    // prologue: stage tile 0
    globalLoad(0);
    ldsStore(0);
    __syncthreads();

    int buf = 0;
    for (int kt = 0; kt < K; kt += 32) {
        const bool hasNext = (kt + 32 < K);
        if (hasNext) globalLoad(kt + 32);          // overlap with WMMA below

        // ---- fragments per ISA 16-bit layout: lane<16 -> K 0..7 & 16..23 ----
        const ushort_t* as = As[buf];
        const ushort_t* bs = BsT[buf];
        Frag a[2], b[4];
#pragma unroll
        for (int mt = 0; mt < 2; mt++) {
            int m = wm0 + mt * 16 + ln;
            a[mt].q[0] = *(const u32x4*)&as[m * 40 + k0l];
            a[mt].q[1] = *(const u32x4*)&as[m * 40 + k0l + 16];
        }
#pragma unroll
        for (int nt = 0; nt < 4; nt++) {
            int n = wn0 + nt * 16 + ln;
            b[nt].q[0] = *(const u32x4*)&bs[n * 40 + k0l];
            b[nt].q[1] = *(const u32x4*)&bs[n * 40 + k0l + 16];
        }
#pragma unroll
        for (int mt = 0; mt < 2; mt++)
#pragma unroll
            for (int nt = 0; nt < 4; nt++)
                acc[mt][nt] = wmma_bf16(a[mt], b[nt], acc[mt][nt]);

        if (hasNext) {
            ldsStore(buf ^ 1);                     // other buffer: no race w/ readers
            __syncthreads();                       // single barrier per K-step
            buf ^= 1;
        }
    }

    // ---- epilogue: C layout lane=(n%16, hi half), vgpr r -> row r+8*hi ----
#pragma unroll
    for (int nt = 0; nt < 4; nt++) {
        int n = bn + wn0 + nt * 16 + ln;
        float bv = bias[n];
#pragma unroll
        for (int mt = 0; mt < 2; mt++) {
#pragma unroll
            for (int r = 0; r < 8; r++) {
                int m = bm + wm0 + mt * 16 + r + 8 * hi;
                float c = acc[mt][nt][r] + bv;
                if constexpr (OUT_F32)
                    ((float*)Cptr)[(size_t)m * N + n] = c;
                else
                    ((ushort_t*)Cptr)[(size_t)m * N + n] = f2bf(c);
            }
        }
    }
}

// ---------------------------------------------------------------------------
// Flash attention: block = (b,h) x 64-query tile, 128 threads = 4 waves,
// each wave owns 16 query rows. Streams keys/values in 64-wide chunks.
//   K chunk staged with GLOBAL_LOAD_ASYNC_TO_LDS_B128 (ASYNCcnt),
//   V transposed through VGPRs, S = Q Kt, online softmax via shuffles,
//   P staged through LDS (C-layout -> A-layout), O += P V.
// ---------------------------------------------------------------------------
__global__ __launch_bounds__(128)
void flash_attn(const ushort_t* __restrict__ Qp, const ushort_t* __restrict__ Kp,
                const ushort_t* __restrict__ Vp, ushort_t* __restrict__ Ctx)
{
    constexpr int   Lk = 2048, D = 1024, E = 64;
    constexpr float scale = 0.125f;               // 1/sqrt(64)

    __shared__ ushort_t Ks[64 * 72];              // [key][e], padded
    __shared__ ushort_t VT[64 * 72];              // [e][key], padded
    __shared__ ushort_t Ps[4][16 * 72];           // per-wave P tile [m][key]

    const int tid  = threadIdx.x;
    const int wave = tid >> 5, lane = tid & 31;
    const int ln = lane & 15, hi = lane >> 4;
    const int k0l = hi * 8;

    const int bh = blockIdx.y;
    const int b = bh >> 4, h = bh & 15;
    const int hoff = h * E;
    const int q0 = blockIdx.x * 64;
    const size_t baseBL = (size_t)b * Lk;

    // Q fragments (A-layout) straight from global: head dim is contiguous
    Frag qf[2];
    {
        const ushort_t* Qrow = Qp + (baseBL + q0 + wave * 16 + ln) * D + hoff;
#pragma unroll
        for (int kc = 0; kc < 2; kc++) {
            qf[kc].q[0] = *(const u32x4*)(Qrow + kc * 32 + k0l);
            qf[kc].q[1] = *(const u32x4*)(Qrow + kc * 32 + k0l + 16);
        }
    }

    f32x8 o[4];
#pragma unroll
    for (int t = 0; t < 4; t++) o[t] = (f32x8)0.0f;
    float mI[8], lI[8];
#pragma unroll
    for (int r = 0; r < 8; r++) { mI[r] = -3.0e38f; lI[r] = 0.0f; }

    for (int s0 = 0; s0 < Lk; s0 += 64) {
        __syncthreads();                           // prev chunk's readers done
        // ---- stage K chunk via async copy (no VGPR roundtrip, ASYNCcnt) ----
        // ---- stage V chunk transposed through VGPRs                      ----
#pragma unroll
        for (int t = 0; t < 4; t++) {
            int cid = tid + t * 128;               // 512 x 16B
            int row = cid >> 3, c8 = (cid & 7) * 8;
            unsigned ldsoff = (unsigned)(uintptr_t)&Ks[row * 72 + c8];
            const ushort_t* gsrc = &Kp[(baseBL + s0 + row) * D + hoff + c8];
            asm volatile("global_load_async_to_lds_b128 %0, %1, off"
                         :: "v"(ldsoff), "v"(gsrc) : "memory");
            u32x4 v = *(const u32x4*)&Vp[(baseBL + s0 + row) * D + hoff + c8];
            const ushort_t* vh = (const ushort_t*)&v;
#pragma unroll
            for (int j = 0; j < 8; j++) VT[(c8 + j) * 72 + row] = vh[j];
        }
        // prefetch next chunk (K rows for tid<64, V rows for tid>=64)
        if (s0 + 64 < Lk) {
            const ushort_t* base = (tid < 64) ? Kp : Vp;
            __builtin_prefetch(&base[(baseBL + s0 + 64 + (tid & 63)) * D + hoff], 0, 1);
        }
        asm volatile("s_wait_asynccnt 0x0" ::: "memory");
        __syncthreads();

        // ---- scores: S(16 x 64) for this wave's rows ----
        f32x8 s[4];
#pragma unroll
        for (int nt = 0; nt < 4; nt++) s[nt] = (f32x8)0.0f;
#pragma unroll
        for (int nt = 0; nt < 4; nt++) {
#pragma unroll
            for (int kc = 0; kc < 2; kc++) {
                Frag bfr;                           // B col n = key row: contiguous in Ks
                bfr.q[0] = *(const u32x4*)&Ks[(nt * 16 + ln) * 72 + kc * 32 + k0l];
                bfr.q[1] = *(const u32x4*)&Ks[(nt * 16 + ln) * 72 + kc * 32 + k0l + 16];
                s[nt] = wmma_bf16(qf[kc], bfr, s[nt]);
            }
        }

        // ---- online softmax (per accumulator row r; row lives in a 16-lane half) ----
#pragma unroll
        for (int r = 0; r < 8; r++) {
            float tm = fmaxf(fmaxf(s[0][r], s[1][r]), fmaxf(s[2][r], s[3][r]));
#pragma unroll
            for (int d = 1; d < 16; d <<= 1) tm = fmaxf(tm, __shfl_xor(tm, d));
            float mNew = fmaxf(mI[r], tm * scale);
            float corr = __expf(mI[r] - mNew);
            mI[r] = mNew;
            float rs = 0.0f;
            int prow = (r + 8 * hi) * 72;
#pragma unroll
            for (int nt = 0; nt < 4; nt++) {
                float p = __expf(s[nt][r] * scale - mNew);
                rs += p;
                Ps[wave][prow + nt * 16 + ln] = f2bf(p);
            }
#pragma unroll
            for (int d = 1; d < 16; d <<= 1) rs += __shfl_xor(rs, d);
            lI[r] = lI[r] * corr + rs;
#pragma unroll
            for (int t = 0; t < 4; t++) o[t][r] *= corr;
        }

        // ---- O += P @ V ----
        Frag pf[2];
#pragma unroll
        for (int kc = 0; kc < 2; kc++) {
            pf[kc].q[0] = *(const u32x4*)&Ps[wave][ln * 72 + kc * 32 + k0l];
            pf[kc].q[1] = *(const u32x4*)&Ps[wave][ln * 72 + kc * 32 + k0l + 16];
        }
#pragma unroll
        for (int t = 0; t < 4; t++) {
#pragma unroll
            for (int kc = 0; kc < 2; kc++) {
                Frag vb;                            // B col n = e: contiguous in VT
                vb.q[0] = *(const u32x4*)&VT[(t * 16 + ln) * 72 + kc * 32 + k0l];
                vb.q[1] = *(const u32x4*)&VT[(t * 16 + ln) * 72 + kc * 32 + k0l + 16];
                o[t] = wmma_bf16(pf[kc], vb, o[t]);
            }
        }
    }

    // ---- normalize + write context (bf16, feeds output projection) ----
#pragma unroll
    for (int t = 0; t < 4; t++) {
#pragma unroll
        for (int r = 0; r < 8; r++) {
            float inv = 1.0f / lI[r];
            size_t grow = baseBL + q0 + wave * 16 + r + 8 * hi;
            Ctx[grow * D + hoff + t * 16 + ln] = f2bf(o[t][r] * inv);
        }
    }
}

extern "C" void kernel_launch(void* const* d_in, const int* in_sizes, int n_in,
                              void* d_out, int out_size, void* d_ws, size_t ws_size,
                              hipStream_t stream)
{
    const float* q  = (const float*)d_in[0];
    const float* k  = (const float*)d_in[1];
    const float* v  = (const float*)d_in[2];
    const float* Wq = (const float*)d_in[3];
    const float* bq = (const float*)d_in[4];
    const float* Wk = (const float*)d_in[5];
    const float* bk = (const float*)d_in[6];
    const float* Wv = (const float*)d_in[7];
    const float* bv = (const float*)d_in[8];
    const float* Wo = (const float*)d_in[9];
    const float* bo = (const float*)d_in[10];

    const int M = 4 * 2048, N = 1024, K = 1024;   // B*L, D, D
    const size_t elems = (size_t)M * N;

    // workspace: Qp | Kp | Vp | Ctx, all bf16 [B*L, D]  (4 x 16 MB = 64 MB)
    ushort_t* Qp  = (ushort_t*)d_ws;
    ushort_t* Kp  = Qp + elems;
    ushort_t* Vp  = Kp + elems;
    ushort_t* Ctx = Vp + elems;

    dim3 gg(N / 128, M / 128);
    gemm_wmma<false, false><<<gg, 256, 0, stream>>>(q, Wq, bq, Qp, M, N, K);
    gemm_wmma<false, false><<<gg, 256, 0, stream>>>(k, Wk, bk, Kp, M, N, K);
    gemm_wmma<false, false><<<gg, 256, 0, stream>>>(v, Wv, bv, Vp, M, N, K);

    flash_attn<<<dim3(2048 / 64, 4 * 16), 128, 0, stream>>>(Qp, Kp, Vp, Ctx);

    gemm_wmma<true, true><<<gg, 256, 0, stream>>>(Ctx, Wo, bo, (float*)d_out, M, N, K);
}
